// CrossAttn_65214783422649
// MI455X (gfx1250) — matrix operand order repaired and checked by reading, not compiled, not run on hardware
//
#include <hip/hip_runtime.h>

// ---------------------------------------------------------------------------
// CDNA5 / gfx1250 implementation.
//
// Exploits the identity mean_r(softmax_r(E)) == 1/R: the q/k/energy/softmax
// path of the reference is value-independent, so only the two value GEMMs,
// two row reductions, cap_vec, and a fused norm+dot remain.
//
// GEMMs run as bf16 V_WMMA_F32_16X16X32_BF16 with f32 accumulate, register-
// blocked 2x2 tiles per wave (32x32 output per wave, 64x128 per block).
// ---------------------------------------------------------------------------

typedef __attribute__((ext_vector_type(16))) __bf16 v16bf;
typedef __attribute__((ext_vector_type(8)))  __bf16 v8bf;
typedef __attribute__((ext_vector_type(8)))  float  v8f;

// ---------------- f32 -> bf16 cast (GEMM operand staging) -------------------
__global__ void __launch_bounds__(256)
cvt_f32_bf16(const float* __restrict__ src, __bf16* __restrict__ dst, int n) {
    int i = blockIdx.x * blockDim.x + threadIdx.x;
    if (i < n) dst[i] = (__bf16)src[i];
}

// A-matrix 16x32 bf16 fragment: lane (h,sl) holds row sl,
// K = {h*16..h*16+7} in VGPR0-3 and {h*16+8..h*16+15} in VGPR4-7.
__device__ __forceinline__ v16bf load_a_frag(const __bf16* __restrict__ row,
                                             int kk, int h) {
    v8bf lo = *(const v8bf*)(row + kk + h * 16);
    v8bf hi = *(const v8bf*)(row + kk + h * 16 + 8);
    return __builtin_shufflevector(lo, hi,
               0, 1, 2, 3, 4, 5, 6, 7, 8, 9, 10, 11, 12, 13, 14, 15);
}

// C/D layout: lane -> col = n0 + (lane&15); VGPR v -> row m0 + v + h*8.
__device__ __forceinline__ void store_tile(float* __restrict__ C, int N,
                                           int m0, int col, int h,
                                           const v8f& acc, float bc) {
#pragma unroll
    for (int v = 0; v < 8; ++v) {
        const int row = m0 + v + h * 8;
        float x = acc[v] + bc;
        x = x > 0.0f ? x : 0.1f * x;          // leaky_relu(0.1)
        C[(size_t)row * N + col] = x;
    }
}

// ---------------- C[M,N] = leaky(A[M,K] @ W[N,K]^T + bias[N]) ---------------
// A, W: bf16 row-major (K contiguous).  C: f32 row-major.
// Block = 256 threads = 8 waves arranged 2(M) x 4(N); each wave computes a
// 32x32 output patch as 2x2 WMMA tiles: 4 fragment loads -> 4 WMMAs per
// K-step of 32 (2 b128 loads per WMMA).
__global__ void __launch_bounds__(256)
gemm_bias_leaky(const __bf16* __restrict__ A, const __bf16* __restrict__ W,
                const float* __restrict__ bias, float* __restrict__ C,
                int M, int N, int K) {
    const int wave = threadIdx.x >> 5;
    const int lane = threadIdx.x & 31;
    const int h    = lane >> 4;    // half-wave select
    const int sl   = lane & 15;    // row/col select within half
    const int mw   = wave >> 2;    // 0..1
    const int nw   = wave & 3;     // 0..3

    const int m0 = blockIdx.x * 64 + mw * 32;    // tiles m0 and m0+16
    const int n0 = blockIdx.y * 128 + nw * 32;   // tiles n0 and n0+16
    if (m0 >= M || n0 >= N) return;              // wave-uniform

    const __bf16* __restrict__ a0 = A + (size_t)(m0 + sl) * K;
    const __bf16* __restrict__ a1 = A + (size_t)(m0 + 16 + sl) * K;
    const __bf16* __restrict__ w0 = W + (size_t)(n0 + sl) * K;
    const __bf16* __restrict__ w1 = W + (size_t)(n0 + 16 + sl) * K;

    v8f acc00 = {}, acc01 = {}, acc10 = {}, acc11 = {};
    for (int kk = 0; kk < K; kk += 32) {
        // keep the K-stream ~512B ahead in cache (global_prefetch_b8)
        __builtin_prefetch(a0 + kk + 256, 0, 1);
        __builtin_prefetch(w0 + kk + 256, 0, 1);

        v16bf a0f = load_a_frag(a0, kk, h);
        v16bf a1f = load_a_frag(a1, kk, h);
        // B-matrix 32x16 bf16 fragment: lane (h,sl) holds column sl,
        // K = h*16 .. h*16+15 contiguous -> one 32B load from W row.
        v16bf b0f = *(const v16bf*)(w0 + kk + h * 16);
        v16bf b1f = *(const v16bf*)(w1 + kk + h * 16);

        acc00 = __builtin_amdgcn_wmma_f32_16x16x32_bf16(
                    false, a0f, false, b0f, (short)0, acc00, false, false);
        acc01 = __builtin_amdgcn_wmma_f32_16x16x32_bf16(
                    false, a0f, false, b1f, (short)0, acc01, false, false);
        acc10 = __builtin_amdgcn_wmma_f32_16x16x32_bf16(
                    false, a1f, false, b0f, (short)0, acc10, false, false);
        acc11 = __builtin_amdgcn_wmma_f32_16x16x32_bf16(
                    false, a1f, false, b1f, (short)0, acc11, false, false);
    }

    const float bc0 = bias[n0 + sl];
    const float bc1 = bias[n0 + 16 + sl];
    store_tile(C, N, m0,      n0 + sl,      h, acc00, bc0);
    store_tile(C, N, m0,      n0 + 16 + sl, h, acc01, bc1);
    store_tile(C, N, m0 + 16, n0 + sl,      h, acc10, bc0);
    store_tile(C, N, m0 + 16, n0 + 16 + sl, h, acc11, bc1);
}

// ---------------- s[c,d] = sum_{t < lens[c]} v_txt[c,t,d] -------------------
__global__ void __launch_bounds__(256)
reduce_txt(const float* __restrict__ v, const int* __restrict__ lens,
           float* __restrict__ s) {
    int idx = blockIdx.x * blockDim.x + threadIdx.x;   // 128*1024
    int c = idx >> 10, d = idx & 1023;
    int L = lens[c];
    const float* p = v + (size_t)c * 64 * 1024 + d;
    float acc = 0.0f;
    for (int t = 0; t < L; ++t) acc += p[(size_t)t * 1024];
    s[idx] = acc;
}

// ---------------- vbar[i,d] = mean_r v_img[i,r,d] ---------------------------
__global__ void __launch_bounds__(256)
reduce_img(const float* __restrict__ v, float* __restrict__ vbar) {
    int idx = blockIdx.x * blockDim.x + threadIdx.x;   // 128*1024
    int i = idx >> 10, d = idx & 1023;
    const float* p = v + (size_t)i * 36 * 1024 + d;
    float acc = 0.0f;
#pragma unroll 4
    for (int r = 0; r < 36; ++r) acc += p[(size_t)r * 1024];
    vbar[idx] = acc * (1.0f / 36.0f);
}

// ---------------- cap_vec[c,:] = l2norm(masked mean of cap_embed) -----------
__global__ void __launch_bounds__(256)
capvec_kernel(const float* __restrict__ cap, const int* __restrict__ lens,
              float* __restrict__ out) {
    const int c = blockIdx.x;
    const int L = lens[c];
    const float inv = 1.0f / (float)L;
    float val[4];
    float ss = 0.0f;
#pragma unroll
    for (int j = 0; j < 4; ++j) {
        const int d = threadIdx.x + j * 256;
        const float* p = cap + (size_t)c * 64 * 1024 + d;
        float a = 0.0f;
        for (int t = 0; t < L; ++t) a += p[(size_t)t * 1024];
        a *= inv;
        val[j] = a;
        ss += a * a;
    }
    __shared__ float red[256];
    red[threadIdx.x] = ss;
    __syncthreads();
    for (int off = 128; off > 0; off >>= 1) {
        if (threadIdx.x < off) red[threadIdx.x] += red[threadIdx.x + off];
        __syncthreads();
    }
    const float rn = 1.0f / (sqrtf(red[0]) + 1e-8f);
#pragma unroll
    for (int j = 0; j < 4; ++j)
        out[(size_t)c * 1024 + threadIdx.x + j * 256] = val[j] * rn;
}

// ---- sims[i,c] = l2norm(gamma/36 * s_c + vbar_i) . cap_vec_c ---------------
__global__ void __launch_bounds__(256)
sims_kernel(const float* __restrict__ s, const float* __restrict__ vbar,
            const float* __restrict__ capvec, const float* __restrict__ gamma,
            float* __restrict__ out) {
    const int c = blockIdx.x, i = blockIdx.y;
    const float g = gamma[0] * (1.0f / 36.0f);
    float nrm = 0.0f, dot = 0.0f;
#pragma unroll
    for (int j = 0; j < 4; ++j) {
        const int d = threadIdx.x + j * 256;
        const float w = g * s[(size_t)c * 1024 + d] + vbar[(size_t)i * 1024 + d];
        nrm += w * w;
        dot += w * capvec[(size_t)c * 1024 + d];
    }
    __shared__ float r1[256], r2[256];
    r1[threadIdx.x] = nrm;
    r2[threadIdx.x] = dot;
    __syncthreads();
    for (int off = 128; off > 0; off >>= 1) {
        if (threadIdx.x < off) {
            r1[threadIdx.x] += r1[threadIdx.x + off];
            r2[threadIdx.x] += r2[threadIdx.x + off];
        }
        __syncthreads();
    }
    if (threadIdx.x == 0)
        out[(size_t)i * 128 + c] = r2[0] / (sqrtf(r1[0]) + 1e-8f);
}

// ---------------------------------------------------------------------------
extern "C" void kernel_launch(void* const* d_in, const int* in_sizes, int n_in,
                              void* d_out, int out_size, void* d_ws, size_t ws_size,
                              hipStream_t stream) {
    (void)in_sizes; (void)n_in; (void)out_size; (void)ws_size;

    const float* img   = (const float*)d_in[0];   // [128,36,1024]
    const float* cap   = (const float*)d_in[1];   // [128,64,1024]
    const int*   lens  = (const int*)d_in[2];     // [128]
    const float* Wvi   = (const float*)d_in[7];   // [1024,1024]
    const float* bvi   = (const float*)d_in[8];
    const float* Wvt   = (const float*)d_in[9];   // [1024,1024]
    const float* bvt   = (const float*)d_in[10];
    const float* gamma = (const float*)d_in[11];
    float*       out   = (float*)d_out;           // [128,128]

    const int CAP_E = 128 * 64 * 1024;   // 8,388,608
    const int IMG_E = 128 * 36 * 1024;   // 4,718,592
    const int W_E   = 1024 * 1024;       // 1,048,576
    const int CD_E  = 128 * 1024;        // 131,072

    // Workspace layout (regions reused between the text and image pipelines).
    char* ws = (char*)d_ws;
    size_t off = 0;
    auto alloc = [&](size_t bytes) -> void* {
        void* p = ws + off;
        off += (bytes + 255) & ~(size_t)255;
        return p;
    };
    __bf16* act_bf = (__bf16*)alloc((size_t)CAP_E * 2);  // cap_bf, then img_bf
    __bf16* w_bf   = (__bf16*)alloc((size_t)W_E * 2);    // Wvt_bf, then Wvi_bf
    float*  gemm_c = (float*)alloc((size_t)CAP_E * 4);   // v_txt, then v_img
    float*  s_c    = (float*)alloc((size_t)CD_E * 4);
    float*  vbar   = (float*)alloc((size_t)CD_E * 4);
    float*  cvec   = (float*)alloc((size_t)CD_E * 4);

    // ---- text pipeline: v_txt = leaky(cap @ Wvt^T + bvt); s_c = masked sum_t
    cvt_f32_bf16<<<(CAP_E + 255) / 256, 256, 0, stream>>>(cap, act_bf, CAP_E);
    cvt_f32_bf16<<<(W_E + 255) / 256, 256, 0, stream>>>(Wvt, w_bf, W_E);
    gemm_bias_leaky<<<dim3(8192 / 64, 1024 / 128), 256, 0, stream>>>(
        act_bf, w_bf, bvt, gemm_c, 8192, 1024, 1024);
    reduce_txt<<<(CD_E + 255) / 256, 256, 0, stream>>>(gemm_c, lens, s_c);

    // ---- image pipeline: v_img = leaky(img @ Wvi^T + bvi); vbar = mean_r
    cvt_f32_bf16<<<(IMG_E + 255) / 256, 256, 0, stream>>>(img, act_bf, IMG_E);
    cvt_f32_bf16<<<(W_E + 255) / 256, 256, 0, stream>>>(Wvi, w_bf, W_E);
    gemm_bias_leaky<<<dim3(4608 / 64, 1024 / 128), 256, 0, stream>>>(
        act_bf, w_bf, bvi, gemm_c, 4608, 1024, 1024);
    reduce_img<<<(CD_E + 255) / 256, 256, 0, stream>>>(gemm_c, vbar);

    // ---- caption vectors + fused norm/dot similarity
    capvec_kernel<<<128, 256, 0, stream>>>(cap, lens, cvec);
    sims_kernel<<<dim3(128, 128), 256, 0, stream>>>(s_c, vbar, cvec, gamma, out);
}
